// GNN2StateEncoder_38139309588793
// MI455X (gfx1250) — matrix operand order repaired and checked by reading, not compiled, not run on hardware
//
#include <hip/hip_runtime.h>
#include <hip/hip_bf16.h>
#include <math.h>

// ---------- problem constants ----------
#define BB      2
#define NN      100000
#define EE      400000
#define NODE_DIM 32
#define NUM_FEAT 32
#define H0      128
#define H1      64
#define DD      64
#define K2      128   // 2*D

typedef __attribute__((ext_vector_type(16))) _Float16 v16h;
typedef __attribute__((ext_vector_type(8)))  float    v8f;
typedef unsigned int u32x4 __attribute__((ext_vector_type(4)));
typedef int          i32x8 __attribute__((ext_vector_type(8)));
typedef int          i32x4 __attribute__((ext_vector_type(4)));

#if __has_builtin(__builtin_amdgcn_tensor_load_to_lds) && \
    __has_builtin(__builtin_amdgcn_s_wait_tensorcnt)
#define HAVE_TDM 1
#else
#define HAVE_TDM 0
#endif

// Hardware tanh (V_TANH_F32 on gfx1250); branchless exp2/rcp fallback.
static __device__ __forceinline__ float fast_tanh(float x) {
#if __has_builtin(__builtin_amdgcn_tanhf)
    return __builtin_amdgcn_tanhf(x);
#elif __has_builtin(__builtin_amdgcn_tanh_f32)
    return __builtin_amdgcn_tanh_f32(x);
#else
    // tanh(x) = 1 - 2/(e^(2x)+1); saturates correctly at +-inf
    float e = __builtin_amdgcn_exp2f(x * 2.885390081777927f);   // e^(2x)
    return 1.0f - 2.0f * __builtin_amdgcn_rcpf(e + 1.0f);
#endif
}

// Build a 16-half A/B fragment from two 16B-aligned groups of 8 halves.
static __device__ __forceinline__ v16h ld_frag_pair(const _Float16* p0, const _Float16* p1) {
    union { v16h v; uint4 q[2]; } u;
    u.q[0] = *reinterpret_cast<const uint4*>(p0);
    u.q[1] = *reinterpret_cast<const uint4*>(p1);
    return u.v;
}

// ---------------------------------------------------------------------------
// Kernel 0: prep — transpose Wn (32x64) -> WnT f16 (64x32, n-major),
//                  transpose We (128x64) -> WeT f16 (64x128, n-major),
//                  zero the mean accumulators.
// ---------------------------------------------------------------------------
__global__ void prep_kernel(const float* __restrict__ Wn,
                            const float* __restrict__ We,
                            _Float16* __restrict__ WnT,
                            _Float16* __restrict__ WeT,
                            float* __restrict__ col_sums,   // [B*64]
                            float* __restrict__ cnt)        // [B]
{
    int tid = threadIdx.x;
    for (int i = tid; i < DD * NODE_DIM; i += blockDim.x) {
        int n = i / NODE_DIM, k = i % NODE_DIM;
        WnT[i] = (_Float16)Wn[k * DD + n];
    }
    for (int i = tid; i < DD * K2; i += blockDim.x) {
        int n = i / K2, k = i % K2;
        WeT[i] = (_Float16)We[k * DD + n];
    }
    if (tid < BB * DD) col_sums[tid] = 0.0f;
    if (tid < BB)      cnt[tid] = 0.0f;
}

// ---------------------------------------------------------------------------
// Kernel 1: h_nodes = tanh(node_feature @ Wn + bn), stored as f16.
// One 16-row tile per wave; K=32 (one WMMA per 16-col n-tile).
// ---------------------------------------------------------------------------
__global__ __launch_bounds__(256)
void node_kernel(const float* __restrict__ nf,        // [B*N, 32] f32
                 const _Float16* __restrict__ WnT,    // [64, 32] n-major f16
                 const float* __restrict__ bn,        // [64]
                 _Float16* __restrict__ hn)           // [B*N, 64] f16
{
    const int NT = (BB * NN) / 16;   // 12500
    int tile = (int)((blockIdx.x * blockDim.x + threadIdx.x) >> 5);
    if (tile >= NT) return;          // wave-uniform exit; no barriers below
    int lane = threadIdx.x & 31;
    int r  = lane & 15;
    int hi = lane >> 4;              // K sub-group select
    int base = tile * 16;

    // A fragment (16x32 f16): lane holds row M=r, K = {hi*8..hi*8+7, 16+hi*8..+7}
    const float* arow = nf + (size_t)(base + r) * NODE_DIM;
    float4 f0 = *reinterpret_cast<const float4*>(arow + hi * 8);
    float4 f1 = *reinterpret_cast<const float4*>(arow + hi * 8 + 4);
    float4 f2 = *reinterpret_cast<const float4*>(arow + 16 + hi * 8);
    float4 f3 = *reinterpret_cast<const float4*>(arow + 16 + hi * 8 + 4);
    v16h a;
    a[0]=(_Float16)f0.x; a[1]=(_Float16)f0.y; a[2]=(_Float16)f0.z; a[3]=(_Float16)f0.w;
    a[4]=(_Float16)f1.x; a[5]=(_Float16)f1.y; a[6]=(_Float16)f1.z; a[7]=(_Float16)f1.w;
    a[8]=(_Float16)f2.x; a[9]=(_Float16)f2.y; a[10]=(_Float16)f2.z; a[11]=(_Float16)f2.w;
    a[12]=(_Float16)f3.x; a[13]=(_Float16)f3.y; a[14]=(_Float16)f3.z; a[15]=(_Float16)f3.w;

    #pragma unroll
    for (int nt = 0; nt < 4; ++nt) {
        // B fragment (32x16): lane holds col N = nt*16+r, K = {hi*8..hi*8+7, 16+hi*8..+7}
        const _Float16* bp = WnT + (size_t)(nt * 16 + r) * NODE_DIM + hi * 8;
        v16h bfrag = ld_frag_pair(bp, bp + 16);
        v8f acc = {};
        acc = __builtin_amdgcn_wmma_f32_16x16x32_f16(
                false, a, false, bfrag, (short)0, acc, false, false);
        int col = nt * 16 + r;
        float bias = bn[col];
        #pragma unroll
        for (int v = 0; v < 8; ++v) {
            int row = base + v + hi * 8;   // C layout: VGPR v -> M = v (+8 for hi half)
            hn[(size_t)row * DD + col] = (_Float16)fast_tanh(acc[v] + bias);
        }
    }
}

// ---------------------------------------------------------------------------
// Kernel 2: edge MLP. One 16-edge tile per wave, 8 waves/block.
// e12 = tanh([h1,h2]@We + be), e21 = tanh([h2,h1]@We + be)  (chunk-swap trick)
// We (16 KB) staged into LDS via the Tensor Data Mover when available.
// ---------------------------------------------------------------------------
__global__ __launch_bounds__(256)
void edge_kernel(const int* __restrict__ ei,             // [B,E,2]
                 const unsigned char* __restrict__ mask, // [B,E] bool
                 const _Float16* __restrict__ hn,        // [B*N, 64] f16
                 const _Float16* __restrict__ WeT,       // [64,128] n-major f16
                 const float* __restrict__ be,           // [64]
                 float* __restrict__ out_policy,         // [B,E,64]
                 float* __restrict__ col_sums,           // [B*64]
                 float* __restrict__ cnt)                // [B]
{
    __shared__ _Float16 sWe[DD * K2];   // 16 KB
    __shared__ float sSums[DD];
    __shared__ float sCnt;

    int tid = threadIdx.x;
#if HAVE_TDM
    if (tid == 0) {
        // TDM D# for a 1D copy: 4096 x 1 tile of 4-byte elements (16 KB).
        unsigned long long ga = (unsigned long long)(uintptr_t)WeT;
        unsigned ldsa = (unsigned)(uintptr_t)&sWe[0];   // low 32 bits = LDS offset
        u32x4 g0;
        g0[0] = 1u;                                        // count=1, user mode
        g0[1] = ldsa;                                      // lds_addr [63:32]
        g0[2] = (unsigned)(ga & 0xFFFFFFFFull);            // global_addr[31:0]
        g0[3] = (unsigned)((ga >> 32) & 0x1FFFFFFull)      // global_addr[56:32]
              | (2u << 30);                                // type=2 ("image")
        i32x8 g1;
        g1[0] = (2 << 16);              // data_size = 4 bytes; no flags
        g1[1] = (4096 & 0xFFFF) << 16;  // tensor_dim0[15:0] @ bits 63:48
        g1[2] = (4096 >> 16) | (1 << 16); // tensor_dim0[31:16] | tensor_dim1=1 @ bit 80
        g1[3] = (4096 << 16);           // tile_dim0 = 4096 @ bits 127:112
        g1[4] = 1;                      // tile_dim1 = 1
        g1[5] = 4096;                   // tensor_dim0_stride (low 32)
        g1[6] = 0;
        g1[7] = 0;
        i32x4 gz4 = {0, 0, 0, 0};
        i32x8 gz8 = {0, 0, 0, 0, 0, 0, 0, 0};
        // 6-arg form (amdgpu-toolchain clang): (g0, g1, g2, g3, g4, cpol)
        __builtin_amdgcn_tensor_load_to_lds(g0, g1, gz4, gz4, gz8, 0);
    }
    __builtin_amdgcn_s_wait_tensorcnt(0);  // no-op for waves with TENSORcnt==0
#else
    {   // fallback: stage WeT into LDS (1024 uint4)
        const uint4* src = reinterpret_cast<const uint4*>(WeT);
        uint4* dst = reinterpret_cast<uint4*>(sWe);
        for (int i = tid; i < (DD * K2) / 8; i += blockDim.x) dst[i] = src[i];
    }
#endif
    if (tid < DD) sSums[tid] = 0.0f;
    if (tid == 0) sCnt = 0.0f;
    __syncthreads();

    int tile = blockIdx.x * 8 + (tid >> 5);   // global 16-edge tile
    int lane = tid & 31;
    int r  = lane & 15;
    int hi = lane >> 4;
    int grow = tile * 16;
    int b  = grow / EE;
    int e0 = grow - b * EE;

    // edge endpoints for this lane's A-row
    const int* eip = ei + ((size_t)b * EE + e0 + r) * 2;
    int i0 = eip[0];
    int i1 = eip[1];
    const _Float16* h1 = hn + ((size_t)b * NN + i0) * DD;
    const _Float16* h2 = hn + ((size_t)b * NN + i1) * DD;

    // A fragments for K-chunks 0..3 of [h1,h2]; e21 uses [h2,h1] = chunks kc^2
    v16h a[4];
    #pragma unroll
    for (int c = 0; c < 2; ++c) {
        int kk = c * 32;
        a[c]     = ld_frag_pair(h1 + kk + hi * 8, h1 + kk + 16 + hi * 8);
        a[c + 2] = ld_frag_pair(h2 + kk + hi * 8, h2 + kk + 16 + hi * 8);
    }

    // per-lane mask bytes for the 8 rows it will store (M = hi*8 + v)
    unsigned long long mbits =
        *reinterpret_cast<const unsigned long long*>(mask + (size_t)b * EE + e0 + hi * 8);
    if (r == 0) {   // count masked edges once per 8-row group
        float c = 0.0f;
        #pragma unroll
        for (int j = 0; j < 8; ++j) c += ((mbits >> (8 * j)) & 0xffull) ? 1.0f : 0.0f;
        atomicAdd(&sCnt, c);
    }

    #pragma unroll
    for (int nt = 0; nt < 4; ++nt) {
        v8f accA = {};   // e12
        v8f accB = {};   // e21
        #pragma unroll
        for (int kc = 0; kc < 4; ++kc) {
            const _Float16* bp = &sWe[(size_t)(nt * 16 + r) * K2 + kc * 32 + hi * 8];
            v16h bf = ld_frag_pair(bp, bp + 16);
            accA = __builtin_amdgcn_wmma_f32_16x16x32_f16(
                       false, a[kc], false, bf, (short)0, accA, false, false);
            accB = __builtin_amdgcn_wmma_f32_16x16x32_f16(
                       false, a[kc ^ 2], false, bf, (short)0, accB, false, false);
        }
        int col = nt * 16 + r;
        float bias = be[col];
        float lsum = 0.0f;
        #pragma unroll
        for (int v = 0; v < 8; ++v) {
            float h = 0.5f * (fast_tanh(accA[v] + bias) + fast_tanh(accB[v] + bias));
            bool mk = ((mbits >> (8 * v)) & 0xffull) != 0;
            h = mk ? h : 0.0f;
            lsum += h;
            size_t row = (size_t)b * EE + e0 + v + hi * 8;
            out_policy[row * DD + col] = h;
        }
        atomicAdd(&sSums[col], lsum);
    }

    __syncthreads();
    int bb = (int)(((size_t)blockIdx.x * 128) / EE);   // block-uniform batch
    if (tid < DD) atomicAdd(&col_sums[bb * DD + tid], sSums[tid]);
    if (tid == 0) atomicAdd(&cnt[bb], sCnt);
}

// ---------------------------------------------------------------------------
// Kernel 3: finalize — h_num MLP (B=2), h_mean = sums/cnt, concat with stage.
// ---------------------------------------------------------------------------
__global__ void final_kernel(const float* __restrict__ numerical, // [B,32]
                             const float* __restrict__ stage,     // [B,2]
                             const float* __restrict__ W1, const float* __restrict__ b1,
                             const float* __restrict__ W2, const float* __restrict__ b2,
                             const float* __restrict__ col_sums,
                             const float* __restrict__ cnt,
                             float* __restrict__ out_value)       // [B,130]
{
    __shared__ float hid[BB * H0];
    int tid = threadIdx.x;
    if (tid < BB * H0) {
        int j = tid & (H0 - 1), bb = tid >> 7;
        float s = b1[j];
        for (int k = 0; k < NUM_FEAT; ++k) s += numerical[bb * NUM_FEAT + k] * W1[k * H0 + j];
        hid[bb * H0 + j] = fast_tanh(s);
    }
    __syncthreads();
    if (tid < BB * H1) {
        int j = tid & (H1 - 1), bb = tid >> 6;
        float s = b2[j];
        for (int k = 0; k < H0; ++k) s += hid[bb * H0 + k] * W2[k * H1 + j];
        out_value[bb * 130 + j] = fast_tanh(s);                         // h_num
        out_value[bb * 130 + 64 + j] = col_sums[bb * DD + j] / cnt[bb]; // h_mean
    }
    if (tid < BB * 2) {
        int bb = tid >> 1;
        out_value[bb * 130 + 128 + (tid & 1)] = stage[tid];
    }
}

// ---------------------------------------------------------------------------
extern "C" void kernel_launch(void* const* d_in, const int* in_sizes, int n_in,
                              void* d_out, int out_size, void* d_ws, size_t ws_size,
                              hipStream_t stream) {
    const float* numerical    = (const float*)d_in[0];
    const float* node_feature = (const float*)d_in[1];
    const int*   edge_index   = (const int*)  d_in[2];
    const unsigned char* edge_mask = (const unsigned char*)d_in[3];
    const float* stage        = (const float*)d_in[4];
    const float* W1 = (const float*)d_in[5];
    const float* b1 = (const float*)d_in[6];
    const float* W2 = (const float*)d_in[7];
    const float* b2 = (const float*)d_in[8];
    const float* Wn = (const float*)d_in[9];
    const float* bn = (const float*)d_in[10];
    const float* We = (const float*)d_in[11];
    const float* be = (const float*)d_in[12];

    // workspace layout
    char* ws = (char*)d_ws;
    size_t off = 0;
    _Float16* hn  = (_Float16*)(ws + off); off += (size_t)BB * NN * DD * sizeof(_Float16); // 25.6MB
    off = (off + 255) & ~(size_t)255;
    _Float16* WnT = (_Float16*)(ws + off); off += (size_t)DD * NODE_DIM * sizeof(_Float16);
    off = (off + 255) & ~(size_t)255;
    _Float16* WeT = (_Float16*)(ws + off); off += (size_t)DD * K2 * sizeof(_Float16);
    off = (off + 255) & ~(size_t)255;
    float* col_sums = (float*)(ws + off); off += (size_t)BB * DD * sizeof(float);
    off = (off + 255) & ~(size_t)255;
    float* cnt = (float*)(ws + off);

    float* out_policy = (float*)d_out;
    float* out_value  = (float*)d_out + (size_t)BB * EE * DD;

    prep_kernel<<<1, 256, 0, stream>>>(Wn, We, WnT, WeT, col_sums, cnt);

    const int node_tiles = (BB * NN) / 16;            // 12500
    node_kernel<<<(node_tiles + 7) / 8, 256, 0, stream>>>(node_feature, WnT, bn, hn);

    const int edge_blocks = (BB * EE) / 128;          // 6250
    edge_kernel<<<edge_blocks, 256, 0, stream>>>(edge_index, edge_mask, hn, WeT, be,
                                                 out_policy, col_sums, cnt);

    final_kernel<<<1, 256, 0, stream>>>(numerical, stage, W1, b1, W2, b2,
                                        col_sums, cnt, out_value);
}
